// KgAttention_24747601559685
// MI455X (gfx1250) — matrix operand order repaired and checked by reading, not compiled, not run on hardware
//
#include <hip/hip_runtime.h>
#include <hip/hip_bf16.h>

#define N_ENT   100000
#define EMB     128
#define N_EDGES 600000
#define N_REL   64
#define N_HOPS  3

typedef __attribute__((ext_vector_type(16))) __bf16          v16bf;
typedef __attribute__((ext_vector_type(8)))  __bf16          v8bf;
typedef __attribute__((ext_vector_type(2)))  __bf16          v2bf;
typedef __attribute__((ext_vector_type(8)))  float           v8f;
typedef __attribute__((ext_vector_type(4)))  unsigned int    u32x4;

#if __has_builtin(__builtin_amdgcn_tanhf)
#define TANH(x) __builtin_amdgcn_tanhf(x)
#elif __has_builtin(__builtin_amdgcn_tanh_f32)
#define TANH(x) __builtin_amdgcn_tanh_f32(x)
#else
#define TANH(x) tanhf(x)
#endif

__device__ __forceinline__ unsigned short f2bf(float f) {
    unsigned x = __float_as_uint(f);
    return (unsigned short)((x + 0x7FFFu + ((x >> 16) & 1u)) >> 16);
}
// packed f32x2 -> bf16x2 (hardware cvt if available)
__device__ __forceinline__ unsigned pk2bf(float a, float b) {
#if __has_builtin(__builtin_amdgcn_cvt_pk_bf16_f32)
    v2bf r = __builtin_amdgcn_cvt_pk_bf16_f32(a, b);
    return __builtin_bit_cast(unsigned, r);
#else
    return (unsigned)f2bf(a) | ((unsigned)f2bf(b) << 16);
#endif
}
// monotone float <-> uint key for atomicMax-based segment max
__device__ __forceinline__ unsigned fkey(float f) {
    unsigned b = __float_as_uint(f);
    return (b & 0x80000000u) ? ~b : (b | 0x80000000u);
}
__device__ __forceinline__ float kdec(unsigned k) {
    unsigned b = (k & 0x80000000u) ? (k & 0x7fffffffu) : ~k;
    return __uint_as_float(b);
}

struct B32 { u32x4 a, b; };
struct P16 { v8bf x, y; };
__device__ __forceinline__ v16bf ld16bf(const unsigned short* p) {
    B32 t;
    t.a = *(const u32x4*)p;
    t.b = *(const u32x4*)(p + 8);
    return __builtin_bit_cast(v16bf, t);
}
__device__ __forceinline__ v8bf ld8bf(const unsigned short* p) {
    u32x4 t = *(const u32x4*)p;
    return __builtin_bit_cast(v8bf, t);
}

// ---------------- f32 -> bf16 mirror (2 elems/thread) ----------------
extern "C" __global__ void k_cvt(const float* __restrict__ src, unsigned* __restrict__ dst, int n2) {
    const int i = blockIdx.x * blockDim.x + threadIdx.x; // one per pair
    if (i < n2) dst[i] = pk2bf(src[2 * i], src[2 * i + 1]);
}

// ---------------- hop init: zero accumulators + segment stats ----------------
extern "C" __global__ void k_init(float* __restrict__ dst, unsigned* __restrict__ smax,
                                  float* __restrict__ ssum, float* __restrict__ cnt) {
    const int i = blockIdx.x * blockDim.x + threadIdx.x;
    if (i < N_ENT * EMB) dst[i] = 0.0f;
    if (i < N_ENT) { smax[i] = 0u; ssum[i] = 0.0f; cnt[i] = 0.0f; }
}

// ---------------- attention scores via WMMA ----------------
// one wave = one 16-edge tile; left = h@Qw, right = tanh((rel*t)@Kw); att = rowdot
extern "C" __global__ __launch_bounds__(128)
void k_att(const unsigned short* __restrict__ aggbf, const float* __restrict__ qw,
           const float* __restrict__ kw, const unsigned short* __restrict__ relbf,
           const int* __restrict__ eidx, const int* __restrict__ etype,
           float* __restrict__ att, unsigned* __restrict__ smax) {
    extern __shared__ unsigned short smem[];
    unsigned short* sQ = smem;              // 128*128 bf16, B-matrix swizzled
    unsigned short* sK = smem + 128 * 128;  // 128*128 bf16

    const int tid = threadIdx.x;
    // pre-swizzle Qw/Kw into exact B-matrix register order:
    // storage s = ((nt*4+kb)*32 + lane)*16 + j ;  N = 16*nt+(lane&15), K = 32*kb+16*(lane>>4)+j
    for (int s = tid; s < 128 * 128; s += 128) {
        const int j = s & 15, ln = (s >> 4) & 31, kb = (s >> 9) & 3, nt = s >> 11;
        const int n = 16 * nt + (ln & 15);
        const int k = 32 * kb + 16 * (ln >> 4) + j;
        sQ[s] = f2bf(qw[k * 128 + n]);
        sK[s] = f2bf(kw[k * 128 + n]);
    }
    __syncthreads();

    const int lane = tid & 31;
    const int hi   = lane >> 4;   // half-wave (A/C layout split)
    const int mrow = lane & 15;   // A-matrix row this lane owns
    const int ntile_tot = (N_EDGES + 15) / 16;

    for (int tile = blockIdx.x * 4 + (tid >> 5); tile < ntile_tot; tile += gridDim.x * 4) {
        const int e0 = tile * 16;
        int ee = e0 + mrow; if (ee >= N_EDGES) ee = N_EDGES - 1;
        const int h = eidx[ee];
        const int t = eidx[N_EDGES + ee];
        const int r = etype[ee];
        const unsigned short* hb = aggbf + (size_t)h * EMB;
        const unsigned short* tb = aggbf + (size_t)t * EMB;
        const unsigned short* rb = relbf + (size_t)r * EMB;

        // A tiles (16x32 bf16 x 4 K-blocks), this code's contiguous k order (not the ISA table order):
        // elem j -> K = 32*kb + 16*hi + j (contiguous, matches B staging)
        v16bf aL[4], aR[4];
#pragma unroll
        for (int kb = 0; kb < 4; kb++) {
            const int k0 = 32 * kb + 16 * hi;
            aL[kb] = ld16bf(hb + k0);                 // one contiguous 16-elem run (k0..k0+15)
            P16 p;
            p.x = ld8bf(tb + k0)      * ld8bf(rb + k0);       // packed bf16 mul
            p.y = ld8bf(tb + k0 + 8) * ld8bf(rb + k0 + 8);
            aR[kb] = __builtin_bit_cast(v16bf, p);
        }

        float part[8] = {0, 0, 0, 0, 0, 0, 0, 0};
#pragma unroll
        for (int nt = 0; nt < 8; nt++) {
            v8f accL = {0, 0, 0, 0, 0, 0, 0, 0};
            v8f accR = {0, 0, 0, 0, 0, 0, 0, 0};
#pragma unroll
            for (int kb = 0; kb < 4; kb++) {
                const int off = (((nt * 4 + kb) * 32) + lane) * 16;
                const v16bf bq = ld16bf(sQ + off);
                const v16bf bk = ld16bf(sK + off);
                accL = __builtin_amdgcn_wmma_f32_16x16x32_bf16(
                    false, aL[kb], false, bq, (short)0, accL, false, false);
                accR = __builtin_amdgcn_wmma_f32_16x16x32_bf16(
                    false, aR[kb], false, bk, (short)0, accR, false, false);
            }
#pragma unroll
            for (int q = 0; q < 8; q++) part[q] += accL[q] * TANH(accR[q]);
        }
        // reduce the 16 lanes of each half (C layout: lane holds rows 8*hi..8*hi+7, one col)
#pragma unroll
        for (int d = 1; d < 16; d <<= 1) {
#pragma unroll
            for (int q = 0; q < 8; q++) part[q] += __shfl_xor(part[q], d, 32);
        }
        if (mrow == 0) {
#pragma unroll
            for (int q = 0; q < 8; q++) {
                const int em = e0 + 8 * hi + q;
                if (em < N_EDGES) {
                    att[em] = part[q];
                    atomicMax(smax + eidx[em], fkey(part[q]));
                }
            }
        }
    }
}

// ---------------- exp + segment sum/count ----------------
extern "C" __global__ void k_expsum(const int* __restrict__ eidx, float* __restrict__ att,
                                    const unsigned* __restrict__ smax, float* __restrict__ ssum,
                                    float* __restrict__ cnt) {
    const int e = blockIdx.x * blockDim.x + threadIdx.x;
    if (e >= N_EDGES) return;
    const int h = eidx[e];
    const float ev = expf(att[e] - kdec(smax[h]));
    att[e] = ev;
    atomicAdd(ssum + h, ev);
    atomicAdd(cnt + h, 1.0f);
}

// ---------------- weighted message scatter (1 wave/edge, 4 dims/lane) ----------------
extern "C" __global__ void k_scatter(const float* __restrict__ agg, const float* __restrict__ relw,
                                     const int* __restrict__ eidx, const int* __restrict__ etype,
                                     const float* __restrict__ ew, const float* __restrict__ ssum,
                                     float* __restrict__ dst) {
    const long gid = (long)blockIdx.x * blockDim.x + threadIdx.x;
    const long e = gid >> 5;
    const int lane = (int)(gid & 31);
    if (e >= N_EDGES) return;
    const int h = eidx[e];
    const int t = eidx[N_EDGES + e];
    const int r = etype[e];
    const float w = ew[e] / ssum[h];
    const int d0 = lane * 4;
    const float4 tv = *(const float4*)(agg + (size_t)t * EMB + d0);
    const float4 rv = *(const float4*)(relw + (size_t)r * EMB + d0);
    float* drow = dst + (size_t)h * EMB + d0;
    atomicAdd(drow + 0, w * tv.x * rv.x);
    atomicAdd(drow + 1, w * tv.y * rv.y);
    atomicAdd(drow + 2, w * tv.z * rv.z);
    atomicAdd(drow + 3, w * tv.w * rv.w);
}

// ---------------- mean, L2-normalize, accumulate output, emit bf16 mirror ----------------
extern "C" __global__ void k_final(const float* __restrict__ ent, float* __restrict__ agg,
                                   const float* __restrict__ cnt, float* __restrict__ out,
                                   unsigned* __restrict__ aggbf32, int first) {
    const long gid = (long)blockIdx.x * blockDim.x + threadIdx.x;
    const long n = gid >> 5;
    const int lane = (int)(gid & 31);
    if (n >= N_ENT) return;
    const float c = fmaxf(cnt[n], 1.0f);
    const int d0 = lane * 4;
    float4 v = *(const float4*)(agg + (size_t)n * EMB + d0);
    v.x /= c; v.y /= c; v.z /= c; v.w /= c;
    float ss = v.x * v.x + v.y * v.y + v.z * v.z + v.w * v.w;
#pragma unroll
    for (int d = 1; d < 32; d <<= 1) ss += __shfl_xor(ss, d, 32);
    const float inv = 1.0f / fmaxf(sqrtf(ss), 1e-12f);
    v.x *= inv; v.y *= inv; v.z *= inv; v.w *= inv;
    *(float4*)(agg + (size_t)n * EMB + d0) = v;  // in-place: becomes next agg (f32)
    uint2 bfp; bfp.x = pk2bf(v.x, v.y); bfp.y = pk2bf(v.z, v.w);
    *(uint2*)(aggbf32 + ((size_t)n * EMB + d0) / 2) = bfp;  // bf16 mirror for next hop's WMMA
    const float4 ev = *(const float4*)(ent + (size_t)n * EMB + d0);
    float4 o;
    if (first) {
        o.x = v.x + ev.x; o.y = v.y + ev.y; o.z = v.z + ev.z; o.w = v.w + ev.w;
    } else {
        const float4 p = *(const float4*)(out + (size_t)n * EMB + d0);
        o.x = p.x + v.x + ev.x; o.y = p.y + v.y + ev.y;
        o.z = p.z + v.z + ev.z; o.w = p.w + v.w + ev.w;
    }
    *(float4*)(out + (size_t)n * EMB + d0) = o;
}

extern "C" void kernel_launch(void* const* d_in, const int* in_sizes, int n_in,
                              void* d_out, int out_size, void* d_ws, size_t ws_size,
                              hipStream_t stream) {
    const float* ent   = (const float*)d_in[0];
    const float* relw  = (const float*)d_in[1];
    const float* qw    = (const float*)d_in[2];
    const float* kw    = (const float*)d_in[3];
    const int*   eidx  = (const int*)d_in[4];
    const int*   etype = (const int*)d_in[5];
    float* out = (float*)d_out;

    float* ws = (float*)d_ws;
    float* agg0 = ws;                            // N_ENT*EMB f32
    float* agg1 = agg0 + (size_t)N_ENT * EMB;    // N_ENT*EMB f32
    float* att  = agg1 + (size_t)N_ENT * EMB;    // N_EDGES f32
    unsigned* smax = (unsigned*)(att + N_EDGES); // N_ENT u32
    float* ssum = (float*)(smax + N_ENT);        // N_ENT f32
    float* cnt  = ssum + N_ENT;                  // N_ENT f32
    unsigned short* aggbf = (unsigned short*)(cnt + N_ENT);            // N_ENT*EMB bf16
    unsigned short* relbf = aggbf + (size_t)N_ENT * EMB;               // N_REL*EMB bf16

    const size_t shmem = 2u * 128u * 128u * sizeof(unsigned short); // 64 KB
    const int att_tiles  = (N_EDGES + 15) / 16;
    const int att_blocks = (att_tiles + 3) / 4;

    // one-time bf16 mirrors: entity table (seed for hop 0) + relation table
    k_cvt<<<(N_ENT * EMB / 2 + 255) / 256, 256, 0, stream>>>(ent, (unsigned*)aggbf, N_ENT * EMB / 2);
    k_cvt<<<(N_REL * EMB / 2 + 255) / 256, 256, 0, stream>>>(relw, (unsigned*)relbf, N_REL * EMB / 2);

    const float* src = ent;
    for (int hop = 0; hop < N_HOPS; hop++) {
        float* dst = (hop & 1) ? agg1 : agg0;
        k_init<<<(N_ENT * EMB + 255) / 256, 256, 0, stream>>>(dst, smax, ssum, cnt);
        k_att<<<att_blocks, 128, shmem, stream>>>(aggbf, qw, kw, relbf, eidx, etype, att, smax);
        k_expsum<<<(N_EDGES + 255) / 256, 256, 0, stream>>>(eidx, att, smax, ssum, cnt);
        k_scatter<<<(N_EDGES + 7) / 8, 256, 0, stream>>>(src, relw, eidx, etype, att, ssum, dst);
        // writes f32 agg in-place AND refreshes the bf16 mirror for the next hop's k_att
        k_final<<<(N_ENT + 7) / 8, 256, 0, stream>>>(ent, dst, cnt, out, aggbf ? (unsigned*)aggbf : nullptr, hop == 0);
        src = dst;
    }
}
